// BiasedMHA_5652176961831
// MI455X (gfx1250) — matrix operand order repaired and checked by reading, not compiled, not run on hardware
//
#include <hip/hip_runtime.h>

typedef __attribute__((ext_vector_type(16))) __bf16 v16b;
typedef __attribute__((ext_vector_type(8)))  __bf16 v8b;
typedef __attribute__((ext_vector_type(8)))  float  v8f;

#define BB 8
#define NN 1024
#define DD 768
#define HH 12
#define HD 64
#define SROW 388   // LDS row stride in floats (16B-aligned fills, modest bank conflicts)

__device__ __forceinline__ v8f wmma_bf16(v16b a, v16b b, v8f c) {
  return __builtin_amdgcn_wmma_f32_16x16x32_bf16(false, a, false, b, (short)0, c, false, false);
}

union frag16 { v16b v; v8b h[2]; };

// ---------------------------------------------------------------- convert
__global__ void cvt_kernel(const float* __restrict__ src, __bf16* __restrict__ dst, int n) {
  for (int i = blockIdx.x * blockDim.x + threadIdx.x; i < n; i += gridDim.x * blockDim.x)
    dst[i] = (__bf16)src[i];
}

// ---------------------------------------------------------------- fused QKVG projection
// out[m,n] = sum_k X[m,k] * W[n,k] + bias[n]; 32x32 tile per wave (2x2 WMMA block).
__global__ __launch_bounds__(256)
void proj_kernel(const __bf16* __restrict__ X,
                 const __bf16* __restrict__ Wq, const __bf16* __restrict__ Wk,
                 const __bf16* __restrict__ Wv, const __bf16* __restrict__ Wg,
                 const float* __restrict__ bq, const float* __restrict__ bk,
                 const float* __restrict__ bv, const float* __restrict__ bg,
                 const float* __restrict__ ndata,
                 __bf16* __restrict__ qb, __bf16* __restrict__ kb,
                 __bf16* __restrict__ vt, float* __restrict__ u) {
  const int lane = threadIdx.x & 31;
  const int g = lane >> 4, r = lane & 15;
  const int wave = (blockIdx.x << 3) + (threadIdx.x >> 5);
  const int mt = wave / 24, nt = wave - mt * 24;   // 256 m-tiles x 24 n-tiles of 32
  const int m0 = mt << 5, n0 = nt << 5;
  const int w = blockIdx.y;

  const __bf16* W    = (w == 0) ? Wq : (w == 1) ? Wk : (w == 2) ? Wv : Wg;
  const float*  bias = (w == 0) ? bq : (w == 1) ? bk : (w == 2) ? bv : bg;

  const __bf16* xrow0 = X + (size_t)(m0 + r) * DD;
  const __bf16* xrow1 = xrow0 + (size_t)16 * DD;
  const __bf16* wrow0 = W + (size_t)(n0 + r) * DD;
  const __bf16* wrow1 = wrow0 + (size_t)16 * DD;

  v8f acc[2][2] = {};
  #pragma unroll 2
  for (int k0 = 0; k0 < DD; k0 += 32) {
    frag16 A0, A1, B0, B1;
    A0.h[0] = *(const v8b*)(xrow0 + k0 + g * 8);
    A0.h[1] = *(const v8b*)(xrow0 + k0 + 16 + g * 8);
    A1.h[0] = *(const v8b*)(xrow1 + k0 + g * 8);
    A1.h[1] = *(const v8b*)(xrow1 + k0 + 16 + g * 8);
    B0.h[0] = *(const v8b*)(wrow0 + k0 + g * 8);
    B0.h[1] = *(const v8b*)(wrow0 + k0 + 16 + g * 8);
    B1.h[0] = *(const v8b*)(wrow1 + k0 + g * 8);
    B1.h[1] = *(const v8b*)(wrow1 + k0 + 16 + g * 8);
    acc[0][0] = wmma_bf16(A0.v, B0.v, acc[0][0]);
    acc[0][1] = wmma_bf16(A0.v, B1.v, acc[0][1]);
    acc[1][0] = wmma_bf16(A1.v, B0.v, acc[1][0]);
    acc[1][1] = wmma_bf16(A1.v, B1.v, acc[1][1]);
  }

  #pragma unroll
  for (int mi = 0; mi < 2; ++mi) {
    const int mb = m0 + mi * 16;
    const int b = mb >> 10, srow0 = mb & 1023;  // 16-row subtile never crosses batch
    #pragma unroll
    for (int ni = 0; ni < 2; ++ni) {
      const v8f a = acc[mi][ni];
      const int n = n0 + ni * 16 + r;
      const float bn = bias[n];
      const int h = n >> 6, d = n & 63;

      if (w == 0) {                    // q, scaled, [B,H,N,HD]
        #pragma unroll
        for (int v = 0; v < 8; ++v)
          qb[(((size_t)(b * HH + h) * NN + srow0 + v + 8 * g) << 6) + d] =
              (__bf16)((a[v] + bn) * 0.125f);
      } else if (w == 1) {             // k, [B,H,N,HD]
        #pragma unroll
        for (int v = 0; v < 8; ++v)
          kb[(((size_t)(b * HH + h) * NN + srow0 + v + 8 * g) << 6) + d] =
              (__bf16)(a[v] + bn);
      } else if (w == 2) {             // v transposed, [B,H,HD,N] (paired stores)
        __bf16* base = vt + (((size_t)(b * HH + h) * HD + d) << 10) + srow0 + 8 * g;
        #pragma unroll
        for (int v = 0; v < 8; v += 2) {
          union { unsigned int ui; __bf16 e[2]; } pk;
          pk.e[0] = (__bf16)(a[v] + bn);
          pk.e[1] = (__bf16)(a[v + 1] + bn);
          *reinterpret_cast<unsigned int*>(base + v) = pk.ui;
        }
      } else {                         // gate: u = ndata * sigmoid(a+bg), fp32
        #pragma unroll
        for (int v = 0; v < 8; ++v) {
          size_t idx = (size_t)(mb + v + 8 * g) * DD + n;
          float z = a[v] + bn;
          u[idx] = ndata[idx] * (1.0f / (1.0f + __expf(-z)));
        }
      }
    }
  }
}

// ---------------------------------------------------------------- flash attention
// One workgroup = one (b, 16-row q-tile); 12 waves = 12 heads sharing the bias
// block via LDS. S^T = K*Q^T so softmax rows are per-lane; ctx^T += V^T * P.
__global__ __launch_bounds__(384)
void attn_kernel(const __bf16* __restrict__ qb, const __bf16* __restrict__ kb,
                 const __bf16* __restrict__ vt, const float* __restrict__ bias,
                 __bf16* __restrict__ ctxb) {
  __shared__ float sb[16 * SROW];
  const int b  = blockIdx.x >> 6;
  const int q0 = (blockIdx.x & 63) << 4;
  const int tid = threadIdx.x;
  const int h = tid >> 5;
  const int lane = tid & 31;
  const int g = lane >> 4, r = lane & 15;

  const __bf16* Qrow  = qb + (((size_t)(b * HH + h) * NN + q0 + r) << 6);
  const __bf16* Kbase = kb + (((size_t)(b * HH + h) * NN) << 6);
  const __bf16* Vbase = vt + (((size_t)(b * HH + h) * HD) << 10);

  frag16 Bq0, Bq1;                 // Q^T B-fragments for d0 = 0, 32
  Bq0.h[0] = *(const v8b*)(Qrow + g * 8);
  Bq0.h[1] = *(const v8b*)(Qrow + 16 + g * 8);
  Bq1.h[0] = *(const v8b*)(Qrow + 32 + g * 8);
  Bq1.h[1] = *(const v8b*)(Qrow + 48 + g * 8);

  v8f ctx[4] = {};
  float M = -1e30f, L = 0.0f;

  for (int k0 = 0; k0 < NN; k0 += 32) {
    // Coalesced stage of bias[b, q0:q0+16, k0:k0+32, :] -> LDS (24 KB)
    #pragma unroll
    for (int it = 0; it < 4; ++it) {
      int f = tid + it * 384;                 // 0..1535 float4s
      int q = f / 96, rem = f - q * 96;
      const float4* gp =
          (const float4*)(bias + ((size_t)(b * NN + q0 + q) * NN + k0) * HH) + rem;
      *(float4*)(&sb[q * SROW + rem * 4]) = *gp;
    }
    __syncthreads();

    // S^T tiles for keys k0..k0+15 (S0) and k0+16..k0+31 (S1)
    frag16 A;
    const __bf16* kr0 = Kbase + (size_t)(k0 + r) * HD;
    const __bf16* kr1 = kr0 + 16 * HD;
    v8f S0 = {}, S1 = {};
    A.h[0] = *(const v8b*)(kr0 + g * 8);      A.h[1] = *(const v8b*)(kr0 + 16 + g * 8);
    S0 = wmma_bf16(A.v, Bq0.v, S0);
    A.h[0] = *(const v8b*)(kr0 + 32 + g * 8); A.h[1] = *(const v8b*)(kr0 + 48 + g * 8);
    S0 = wmma_bf16(A.v, Bq1.v, S0);
    A.h[0] = *(const v8b*)(kr1 + g * 8);      A.h[1] = *(const v8b*)(kr1 + 16 + g * 8);
    S1 = wmma_bf16(A.v, Bq0.v, S1);
    A.h[0] = *(const v8b*)(kr1 + 32 + g * 8); A.h[1] = *(const v8b*)(kr1 + 48 + g * 8);
    S1 = wmma_bf16(A.v, Bq1.v, S1);

    // bias add (lane holds q-column r; reg v holds key v+8g) + tile max
    float tm = -1e30f;
    #pragma unroll
    for (int v = 0; v < 8; ++v) {
      S0[v] += sb[r * SROW + (v + 8 * g) * HH + h];
      S1[v] += sb[r * SROW + (16 + v + 8 * g) * HH + h];
      tm = fmaxf(tm, fmaxf(S0[v], S1[v]));
    }
    __syncthreads();

    tm = fmaxf(tm, __shfl_xor(tm, 16, 32));
    const float newM = fmaxf(M, tm);
    const float scale = __expf(M - newM);
    M = newM;

    v8f P0, P1;
    float ts = 0.0f;
    #pragma unroll
    for (int v = 0; v < 8; ++v) {
      P0[v] = __expf(S0[v] - newM);
      P1[v] = __expf(S1[v] - newM);
      ts += P0[v] + P1[v];
    }
    ts += __shfl_xor(ts, 16, 32);
    L = L * scale + ts;
    #pragma unroll
    for (int dt = 0; dt < 4; ++dt)
      #pragma unroll
      for (int v = 0; v < 8; ++v) ctx[dt][v] *= scale;

    // Rebuild P as a B-fragment (K = 32 keys, N = 16 q) via half-wave swaps
    union { v16b v; __bf16 e[16]; } Pb;
    #pragma unroll
    for (int v = 0; v < 8; ++v) {
      float o0 = __shfl_xor(P0[v], 16, 32);
      float o1 = __shfl_xor(P1[v], 16, 32);
      Pb.e[v]     = (__bf16)((g == 0) ? P0[v] : o1);
      Pb.e[v + 8] = (__bf16)((g == 0) ? o0 : P1[v]);
    }

    // ctx^T[dtile] += V^T(16d x 32k) * P(32k x 16q)
    #pragma unroll
    for (int dt = 0; dt < 4; ++dt) {
      frag16 Av;
      const __bf16* vr = Vbase + (((size_t)(dt * 16 + r)) << 10) + k0;
      Av.h[0] = *(const v8b*)(vr + g * 16);
      Av.h[1] = *(const v8b*)(vr + g * 16 + 8);
      ctx[dt] = wmma_bf16(Av.v, Pb.v, ctx[dt]);
    }
  }

  const float rinv = 1.0f / L;
  __bf16* crow = ctxb + ((size_t)(b * NN) + q0 + r) * DD + h * HD;
  #pragma unroll
  for (int dt = 0; dt < 4; ++dt)
    #pragma unroll
    for (int v = 0; v < 8; v += 2) {
      union { unsigned int ui; __bf16 e[2]; } pk;
      pk.e[0] = (__bf16)(ctx[dt][v] * rinv);
      pk.e[1] = (__bf16)(ctx[dt][v + 1] * rinv);
      *reinterpret_cast<unsigned int*>(crow + dt * 16 + 8 * g + v) = pk.ui;
    }
}

// ---------------------------------------------------------------- output GEMM + epilogue
__global__ __launch_bounds__(256)
void out_kernel(const __bf16* __restrict__ C, const __bf16* __restrict__ Wo,
                const float* __restrict__ bo, const float* __restrict__ u,
                float* __restrict__ out) {
  const int lane = threadIdx.x & 31;
  const int g = lane >> 4, r = lane & 15;
  const int wave = (blockIdx.x << 3) + (threadIdx.x >> 5);
  const int mt = wave / 24, nt = wave - mt * 24;
  const int m0 = mt << 5, n0 = nt << 5;

  const __bf16* xrow0 = C + (size_t)(m0 + r) * DD;
  const __bf16* xrow1 = xrow0 + (size_t)16 * DD;
  const __bf16* wrow0 = Wo + (size_t)(n0 + r) * DD;
  const __bf16* wrow1 = wrow0 + (size_t)16 * DD;

  v8f acc[2][2] = {};
  #pragma unroll 2
  for (int k0 = 0; k0 < DD; k0 += 32) {
    frag16 A0, A1, B0, B1;
    A0.h[0] = *(const v8b*)(xrow0 + k0 + g * 8);
    A0.h[1] = *(const v8b*)(xrow0 + k0 + 16 + g * 8);
    A1.h[0] = *(const v8b*)(xrow1 + k0 + g * 8);
    A1.h[1] = *(const v8b*)(xrow1 + k0 + 16 + g * 8);
    B0.h[0] = *(const v8b*)(wrow0 + k0 + g * 8);
    B0.h[1] = *(const v8b*)(wrow0 + k0 + 16 + g * 8);
    B1.h[0] = *(const v8b*)(wrow1 + k0 + g * 8);
    B1.h[1] = *(const v8b*)(wrow1 + k0 + 16 + g * 8);
    acc[0][0] = wmma_bf16(A0.v, B0.v, acc[0][0]);
    acc[0][1] = wmma_bf16(A0.v, B1.v, acc[0][1]);
    acc[1][0] = wmma_bf16(A1.v, B0.v, acc[1][0]);
    acc[1][1] = wmma_bf16(A1.v, B1.v, acc[1][1]);
  }

  #pragma unroll
  for (int mi = 0; mi < 2; ++mi) {
    const int mb = m0 + mi * 16;
    #pragma unroll
    for (int ni = 0; ni < 2; ++ni) {
      const v8f a = acc[mi][ni];
      const int n = n0 + ni * 16 + r;
      const float bn = bo[n];
      #pragma unroll
      for (int v = 0; v < 8; ++v) {
        size_t idx = (size_t)(mb + v + 8 * g) * DD + n;
        out[idx] = a[v] + bn + u[idx];
      }
    }
  }
}

// ---------------------------------------------------------------- host
extern "C" void kernel_launch(void* const* d_in, const int* in_sizes, int n_in,
                              void* d_out, int out_size, void* d_ws, size_t ws_size,
                              hipStream_t stream) {
  (void)in_sizes; (void)n_in; (void)out_size; (void)ws_size;
  const float* ndata = (const float*)d_in[0];
  const float* abias = (const float*)d_in[1];
  const float* Wq = (const float*)d_in[2];  const float* bq = (const float*)d_in[3];
  const float* Wk = (const float*)d_in[4];  const float* bk = (const float*)d_in[5];
  const float* Wv = (const float*)d_in[6];  const float* bv = (const float*)d_in[7];
  const float* Wg = (const float*)d_in[8];  const float* bg = (const float*)d_in[9];
  const float* Wo = (const float*)d_in[10]; const float* bo = (const float*)d_in[11];
  float* out = (float*)d_out;

  char* ws = (char*)d_ws;
  size_t off = 0;
  auto take = [&](size_t bytes) {
    char* p = ws + off;
    off += (bytes + 255) & ~(size_t)255;
    return p;
  };
  const size_t MND = (size_t)BB * NN * DD;
  __bf16* xb   = (__bf16*)take(MND * 2);
  __bf16* wqb  = (__bf16*)take((size_t)DD * DD * 2);
  __bf16* wkb  = (__bf16*)take((size_t)DD * DD * 2);
  __bf16* wvb  = (__bf16*)take((size_t)DD * DD * 2);
  __bf16* wgb  = (__bf16*)take((size_t)DD * DD * 2);
  __bf16* wob  = (__bf16*)take((size_t)DD * DD * 2);
  __bf16* qbuf = (__bf16*)take(MND * 2);
  __bf16* kbuf = (__bf16*)take(MND * 2);
  __bf16* vtb  = (__bf16*)take(MND * 2);
  float*  ubuf = (float*)take(MND * 4);
  __bf16* ctxb = (__bf16*)take(MND * 2);

  cvt_kernel<<<1024, 256, 0, stream>>>(ndata, xb, (int)MND);
  cvt_kernel<<<256, 256, 0, stream>>>(Wq, wqb, DD * DD);
  cvt_kernel<<<256, 256, 0, stream>>>(Wk, wkb, DD * DD);
  cvt_kernel<<<256, 256, 0, stream>>>(Wv, wvb, DD * DD);
  cvt_kernel<<<256, 256, 0, stream>>>(Wg, wgb, DD * DD);
  cvt_kernel<<<256, 256, 0, stream>>>(Wo, wob, DD * DD);

  dim3 pg(768, 4);    // (256 m-tiles * 24 n-tiles) / 8 waves, 4 projections
  proj_kernel<<<pg, 256, 0, stream>>>(xb, wqb, wkb, wvb, wgb,
                                      bq, bk, bv, bg, ndata,
                                      qbuf, kbuf, vtb, ubuf);

  attn_kernel<<<BB * (NN / 16), 384, 0, stream>>>(qbuf, kbuf, vtb, abias, ctxb);

  out_kernel<<<768, 256, 0, stream>>>(ctxb, wob, bo, ubuf, out);
}
